// MoEFFN_71133248356457
// MI455X (gfx1250) — matrix operand order repaired and checked by reading, not compiled, not run on hardware
//
#include <hip/hip_runtime.h>
#include <hip/hip_bf16.h>
#include <math.h>
#include <stdint.h>

// MoE FFN for MI455X (gfx1250): sparse top-2 expert compute, bf16 WMMA GEMMs
// with fp32 accumulation, fused SiLU, router-weighted scatter.
// v3: A-tile staging uses CDNA5 async global->LDS copies (ASYNCcnt) instead of
//     global->VGPR->LDS round trips; weights pre-converted to bf16; 4
//     accumulators per wave (16x64 tile per wave).

typedef __attribute__((ext_vector_type(16))) __bf16 v16bf;
typedef __attribute__((ext_vector_type(8)))  __bf16 v8bf;
typedef __attribute__((ext_vector_type(8)))  float  v8f;

#define N_TOK 8192
#define C_DIM 1024
#define E_NUM 8
#define F_DIM 2048
#define TOPK  2

__device__ __forceinline__ unsigned short bf16_rne(float f) {
  unsigned u = __builtin_bit_cast(unsigned, f);
  u += 0x7FFFu + ((u >> 16) & 1u);
  return (unsigned short)(u >> 16);
}

// async 16B global->LDS copy; LDS offset = low 32 bits of generic pointer.
__device__ __forceinline__ void async_g2l_b128(void* lds_ptr, const void* gptr) {
  unsigned lds_off = (unsigned)(uintptr_t)lds_ptr;
  asm volatile("global_load_async_to_lds_b128 %0, %1, off"
               :: "v"(lds_off), "v"(gptr) : "memory");
}
__device__ __forceinline__ void wait_async0() {
  asm volatile("s_wait_asynccnt 0x0" ::: "memory");
}

union FragB16 { v16bf v; unsigned short u[16]; v8bf h[2]; };

// ---------------------------------------------------------------- init ------
__global__ void init_kernel(float* __restrict__ out, int n4, int* __restrict__ counts) {
  int i = blockIdx.x * blockDim.x + threadIdx.x;
  if (i < E_NUM) counts[i] = 0;
  if (i < n4) ((float4*)out)[i] = make_float4(0.f, 0.f, 0.f, 0.f);
}

// -------------------------------------------------------------- router ------
__global__ void router_kernel(const float* __restrict__ x, const float* __restrict__ Wr,
                              int* __restrict__ counts, int* __restrict__ tid2,
                              float* __restrict__ tw2) {
  const int t = blockIdx.x;
  const int tid = threadIdx.x;
  __shared__ float sred[E_NUM][256];
  __shared__ float sl[E_NUM];
  float part[E_NUM];
#pragma unroll
  for (int e = 0; e < E_NUM; ++e) part[e] = 0.f;
  const float* xr = x + (size_t)t * C_DIM;
  for (int c = tid; c < C_DIM; c += 256) {
    float xv = xr[c];
#pragma unroll
    for (int e = 0; e < E_NUM; ++e) part[e] += xv * Wr[e * C_DIM + c];
  }
#pragma unroll
  for (int e = 0; e < E_NUM; ++e) sred[e][tid] = part[e];
  __syncthreads();
  if (tid < E_NUM) {
    float s = 0.f;
    for (int j = 0; j < 256; ++j) s += sred[tid][j];
    sl[tid] = s;
  }
  __syncthreads();
  if (tid == 0) {
    int e1 = 0; float v1 = sl[0];
    for (int e = 1; e < E_NUM; ++e) if (sl[e] > v1) { v1 = sl[e]; e1 = e; }
    int e2 = (e1 == 0) ? 1 : 0; float v2 = sl[e2];
    for (int e = 0; e < E_NUM; ++e) if (e != e1 && sl[e] > v2) { v2 = sl[e]; e2 = e; }
    float w1 = 1.f / (1.f + expf(v2 - v1));   // softmax over top-2 (v1 >= v2)
    tid2[t * 2] = e1; tid2[t * 2 + 1] = e2;
    tw2[t * 2] = w1;  tw2[t * 2 + 1] = 1.f - w1;
    atomicAdd(&counts[e1], 1);
    atomicAdd(&counts[e2], 1);
  }
}

// ---------------------------------------------------------------- scan ------
__global__ void scan_kernel(const int* __restrict__ counts, int* __restrict__ offsets,
                            int* __restrict__ cursor) {
  if (threadIdx.x == 0) {
    int off = 0;
    for (int e = 0; e < E_NUM; ++e) { offsets[e] = off; off += counts[e]; }
  }
  if (threadIdx.x < E_NUM) cursor[threadIdx.x] = 0;
}

// -------------------------------------------------------------- assign ------
__global__ void assign_kernel(const int* __restrict__ tid2, const float* __restrict__ tw2,
                              const int* __restrict__ offsets, int* __restrict__ cursor,
                              int* __restrict__ pair_token, float* __restrict__ pair_w) {
  int t = blockIdx.x * blockDim.x + threadIdx.x;
  if (t >= N_TOK) return;
#pragma unroll
  for (int k = 0; k < TOPK; ++k) {
    int e = tid2[t * 2 + k];
    int slot = atomicAdd(&cursor[e], 1);
    int pos = offsets[e] + slot;
    pair_token[pos] = t;
    pair_w[pos] = tw2[t * 2 + k];
  }
}

// -------------------------------------------------- f32 -> bf16 (bulk) ------
__global__ void cvt_kernel(const float* __restrict__ src, unsigned short* __restrict__ dst,
                           int n4) {
  int i = blockIdx.x * blockDim.x + threadIdx.x;
  if (i >= n4) return;
  float4 f = ((const float4*)src)[i];
  ushort4 o;
  o.x = bf16_rne(f.x); o.y = bf16_rne(f.y); o.z = bf16_rne(f.z); o.w = bf16_rne(f.w);
  ((ushort4*)dst)[i] = o;
}

// --------------------------------------- GEMM1: h = x @ W1^T, act=val*silu --
// block: 8 waves, each wave owns a 16x64 tile (4 accumulators). waves 0-3
// compute "val" columns, waves 4-7 the matching "gate" columns; gate goes
// through LDS, val waves fuse SiLU and store bf16 act. Block = 16 tok x 256 f.
#define G1_PITCH 1032  // 1024 + 8 ushorts: shifts rows by 4 LDS banks, 16B-mult
__global__ __launch_bounds__(256) void gemm1_silu_kernel(
    const unsigned short* __restrict__ xbf, const unsigned short* __restrict__ W1bf,
    const int* __restrict__ counts, const int* __restrict__ offsets,
    const int* __restrict__ pair_token, unsigned short* __restrict__ actbuf) {
  const int e = blockIdx.z;
  const int cnt = counts[e];
  const int m0 = blockIdx.x * 16;
  if (m0 >= cnt) return;
  const int off_e = offsets[e];
  const int tid = threadIdx.x;
  __shared__ int sTok[16];
  __shared__ unsigned short sA[16 * G1_PITCH];
  __shared__ float sG[4 * 4 * 16 * 16];     // 4 gate waves x 4 tiles x 16x16
  if (tid < 16) {
    int rr = m0 + tid; if (rr >= cnt) rr = cnt - 1;
    sTok[tid] = pair_token[off_e + rr];
  }
  __syncthreads();
  // stage 16 token rows (bf16) via async global->LDS (ASYNCcnt)
  for (int i = tid * 8; i < 16 * C_DIM; i += 256 * 8) {
    int r = i >> 10, c = i & (C_DIM - 1);
    async_g2l_b128(&sA[r * G1_PITCH + c], &xbf[(size_t)sTok[r] * C_DIM + c]);
  }
  wait_async0();
  __syncthreads();
  const int w = tid >> 5, lane = tid & 31;
  const int n = lane & 15, kh = lane >> 4;
  const int g = w & 3;
  const int colbase = blockIdx.y * 256 + g * 64;
  const int rowbase = (w < 4 ? colbase : F_DIM + colbase) + n;
  const unsigned short* wb =
      W1bf + ((size_t)e * (2 * F_DIM) + rowbase) * C_DIM;
  v8f acc[4] = {};
#pragma unroll 2
  for (int kc = 0; kc < C_DIM / 32; ++kc) {
    const int k0 = kc * 32;
    FragB16 fa;
    fa.h[0] = *(const v8bf*)&sA[n * G1_PITCH + k0 + 8 * kh];
    fa.h[1] = *(const v8bf*)&sA[n * G1_PITCH + k0 + 16 + 8 * kh];
#pragma unroll
    for (int a = 0; a < 4; ++a) {
      FragB16 fb;
      const v8bf* bp = (const v8bf*)(wb + (size_t)(a * 16) * C_DIM + k0);
      fb.h[0] = bp[kh];
      fb.h[1] = bp[2 + kh];
      acc[a] = __builtin_amdgcn_wmma_f32_16x16x32_bf16(false, fa.v, false, fb.v,
                                                       (short)0, acc[a], false, false);
    }
  }
  if (w >= 4) {                      // publish 4 gate tiles
#pragma unroll
    for (int a = 0; a < 4; ++a)
#pragma unroll
      for (int r = 0; r < 8; ++r)
        sG[(((w - 4) * 4 + a) * 16 + r + 8 * kh) * 16 + n] = acc[a][r];
  }
  __syncthreads();
  if (w < 4) {                       // val * silu(gate) -> bf16 act
#pragma unroll
    for (int a = 0; a < 4; ++a) {
      const int fcol = blockIdx.y * 256 + w * 64 + a * 16 + n;
#pragma unroll
      for (int r = 0; r < 8; ++r) {
        int M = r + 8 * kh, row = m0 + M;
        if (row < cnt) {
          float gv = sG[((w * 4 + a) * 16 + M) * 16 + n];
          float av = acc[a][r] * gv / (1.f + expf(-gv));
          actbuf[(size_t)(off_e + row) * F_DIM + fcol] = bf16_rne(av);
        }
      }
    }
  }
}

// ---------------------------- GEMM2: y = act @ W2^T, weighted scatter-add ---
// block: 8 waves x 16x64 tiles = 16 rows x 512 C-columns per block.
#define G2_KC 512
#define G2_PITCH 520   // 512 + 8 ushorts, 16B-multiple pitch
__global__ __launch_bounds__(256) void gemm2_scatter_kernel(
    const unsigned short* __restrict__ actbuf, const unsigned short* __restrict__ W2bf,
    const int* __restrict__ counts, const int* __restrict__ offsets,
    const int* __restrict__ pair_token, const float* __restrict__ pair_w,
    float* __restrict__ out) {
  const int e = blockIdx.z;
  const int cnt = counts[e];
  const int m0 = blockIdx.x * 16;
  if (m0 >= cnt) return;
  const int off_e = offsets[e];
  const int tid = threadIdx.x;
  __shared__ unsigned short sAct[16 * G2_PITCH];
  const int w = tid >> 5, lane = tid & 31;
  const int n = lane & 15, kh = lane >> 4;
  const int cbase = blockIdx.y * 512 + w * 64;
  const unsigned short* wb =
      W2bf + ((size_t)e * C_DIM + cbase + n) * F_DIM;
  v8f acc[4] = {};
  for (int cc = 0; cc < F_DIM / G2_KC; ++cc) {
    const int f0 = cc * G2_KC;
    // async global->LDS staging of the 16xKC act slab
    for (int i = tid * 8; i < 16 * G2_KC; i += 256 * 8) {
      int r = i >> 9, c = i & (G2_KC - 1);
      int row = m0 + r; if (row >= cnt) row = cnt - 1;
      async_g2l_b128(&sAct[r * G2_PITCH + c],
                     &actbuf[(size_t)(off_e + row) * F_DIM + f0 + c]);
    }
    wait_async0();
    __syncthreads();
#pragma unroll 2
    for (int ks = 0; ks < G2_KC / 32; ++ks) {
      const int k0 = ks * 32;
      FragB16 fa;
      fa.h[0] = *(const v8bf*)&sAct[n * G2_PITCH + k0 + 8 * kh];
      fa.h[1] = *(const v8bf*)&sAct[n * G2_PITCH + k0 + 16 + 8 * kh];
#pragma unroll
      for (int a = 0; a < 4; ++a) {
        FragB16 fb;
        const v8bf* bp = (const v8bf*)(wb + (size_t)(a * 16) * F_DIM + f0 + k0);
        fb.h[0] = bp[kh];
        fb.h[1] = bp[2 + kh];
        acc[a] = __builtin_amdgcn_wmma_f32_16x16x32_bf16(false, fa.v, false, fb.v,
                                                         (short)0, acc[a], false, false);
      }
    }
    __syncthreads();
  }
#pragma unroll
  for (int a = 0; a < 4; ++a) {
#pragma unroll
    for (int r = 0; r < 8; ++r) {
      int M = r + 8 * kh, row = m0 + M;
      if (row < cnt) {
        int p = off_e + row;
        atomicAdd(&out[(size_t)pair_token[p] * C_DIM + cbase + a * 16 + n],
                  acc[a][r] * pair_w[p]);
      }
    }
  }
}

// ----------------------------------------------------------------------------
extern "C" void kernel_launch(void* const* d_in, const int* in_sizes, int n_in,
                              void* d_out, int out_size, void* d_ws, size_t ws_size,
                              hipStream_t stream) {
  (void)in_sizes; (void)n_in; (void)out_size; (void)ws_size;
  const float* x  = (const float*)d_in[0];
  const float* Wr = (const float*)d_in[1];
  const float* W1 = (const float*)d_in[2];
  const float* W2 = (const float*)d_in[3];
  float* out = (float*)d_out;

  // workspace layout (~176.3 MB total)
  char* ws = (char*)d_ws;
  int*   counts     = (int*)(ws + 0);
  int*   cursor     = (int*)(ws + 256);
  int*   offsets    = (int*)(ws + 512);
  int*   tid2       = (int*)(ws + 1024);                       // N*2 ints
  float* tw2        = (float*)(ws + 1024 + 65536);             // N*2 floats
  int*   pair_token = (int*)(ws + 1024 + 2 * 65536);           // N*2 ints
  float* pair_w     = (float*)(ws + 1024 + 3 * 65536);         // N*2 floats
  unsigned short* xbf    = (unsigned short*)(ws + 263168);                 // 16 MB
  unsigned short* actbuf = (unsigned short*)(ws + 263168 + (16u << 20));   // 64 MB
  unsigned short* W1bf   = (unsigned short*)(ws + 263168 + (80u << 20));   // 64 MB
  unsigned short* W2bf   = (unsigned short*)(ws + 263168 + (144u << 20));  // 32 MB

  const int NC  = N_TOK * C_DIM;                 // 8.39M
  const int NW1 = E_NUM * 2 * F_DIM * C_DIM;     // 33.55M
  const int NW2 = E_NUM * C_DIM * F_DIM;         // 16.78M

  init_kernel<<<NC / 4 / 256, 256, 0, stream>>>(out, NC / 4, counts);
  router_kernel<<<N_TOK, 256, 0, stream>>>(x, Wr, counts, tid2, tw2);
  scan_kernel<<<1, 32, 0, stream>>>(counts, offsets, cursor);
  assign_kernel<<<N_TOK / 256, 256, 0, stream>>>(tid2, tw2, offsets, cursor,
                                                 pair_token, pair_w);
  cvt_kernel<<<NC / 4 / 256, 256, 0, stream>>>(x, xbf, NC / 4);
  cvt_kernel<<<NW1 / 4 / 256, 256, 0, stream>>>(W1, W1bf, NW1 / 4);
  cvt_kernel<<<NW2 / 4 / 256, 256, 0, stream>>>(W2, W2bf, NW2 / 4);
  gemm1_silu_kernel<<<dim3(N_TOK / 16, F_DIM / 256, E_NUM), 256, 0, stream>>>(
      xbf, W1bf, counts, offsets, pair_token, actbuf);
  gemm2_scatter_kernel<<<dim3(N_TOK / 16, C_DIM / 512, E_NUM), 256, 0, stream>>>(
      actbuf, W2bf, counts, offsets, pair_token, pair_w, out);
}
